// ImageSegmenterDecoderTransformer_12515534701103
// MI455X (gfx1250) — matrix operand order repaired and checked by reading, not compiled, not run on hardware
//
#include <hip/hip_runtime.h>
#include <hip/hip_bf16.h>

#define DSZ   32
#define DIMD  512
#define NHD   8
#define HDD   64
#define NED   4
#define NCD   2
#define BATCH 8
#define LAYERS 2
#define FFD   2048
#define SEQ   (NED + DSZ*DSZ)     // 1028
#define MROWS (BATCH*SEQ)         // 8224 = 514*16
#define NJB   33                  // key blocks of 32
#define SEQP  (NJB*32)            // 1056, padded seq for V^T

typedef __attribute__((ext_vector_type(16))) _Float16 v16h;
typedef __attribute__((ext_vector_type(8)))  float    v8f;
typedef __attribute__((__vector_size__(4 * sizeof(int)))) int v4i_t;
typedef _Float16 half_t;

// Async global->LDS (CDNA5): probe-confirmed to exist; param0 is
// addrspace(1) int4* (global source), param1 addrspace(3) int4* (LDS dest).
#if defined(__has_builtin)
#if __has_builtin(__builtin_amdgcn_global_load_async_to_lds_b128)
#define USE_ASYNC_LDS 1
#endif
#endif
#ifndef USE_ASYNC_LDS
#define USE_ASYNC_LDS 0
#endif

__device__ __forceinline__ void copy32B_to_lds(const half_t* gsrc, half_t* sdst) {
#if USE_ASYNC_LDS
  __builtin_amdgcn_global_load_async_to_lds_b128(
      (__attribute__((address_space(1))) v4i_t*)gsrc,
      (__attribute__((address_space(3))) v4i_t*)sdst, 0, 0);
#else
  *(uint4*)sdst = *(const uint4*)gsrc;
#endif
}
__device__ __forceinline__ void lds_copy_drain() {
#if USE_ASYNC_LDS
#if defined(__has_builtin) && __has_builtin(__builtin_amdgcn_s_wait_asynccnt)
  __builtin_amdgcn_s_wait_asynccnt(0);
#else
  asm volatile("s_wait_asynccnt 0x0" ::: "memory");
#endif
#endif
}

// ---------------------------------------------------------------------------
// WMMA helpers (CDNA5 wave32, V_WMMA_F32_16X16X32_F16)
// ---------------------------------------------------------------------------
__device__ __forceinline__ v8f wmma_f16(v16h a, v16h b, v8f c) {
  return __builtin_amdgcn_wmma_f32_16x16x32_f16(false, a, false, b, (short)0, c,
                                                false, false);
}

// A-matrix 16x32 f16 fragment from row-major [*, lda] source.
//   lane<16:  M=lane,    K = {0..15}
//   lane>=16: M=lane-16, K = {16..31}
__device__ __forceinline__ v16h load_a_frag(const half_t* __restrict__ A, int lda,
                                            int row0, int k0, int maxRow) {
  int lane = threadIdx.x & 31;
  int m = row0 + (lane & 15);
  if (m > maxRow) m = maxRow;
  const half_t* p = A + (size_t)m * lda + k0 + ((lane >> 4) ? 16 : 0);
  v16h a;
#pragma unroll
  for (int v = 0; v < 8; ++v) {
    int k = 2 * v;
    a[2 * v]     = p[k];
    a[2 * v + 1] = p[k + 1];
  }
  return a;
}

// B-matrix 32x16 fragment from a K-contiguous source: per lane, 16 contiguous
// halves at row `n` (lanes 0-15: K lower half; 16-31: K upper half).
__device__ __forceinline__ v16h load_b_frag_kcontig(const half_t* __restrict__ P,
                                                    size_t rowStride, int n,
                                                    int k0) {
  int lane = threadIdx.x & 31;
  const half_t* p = P + (size_t)(n + (lane & 15)) * rowStride + k0 +
                    ((lane >> 4) ? 16 : 0);
  v16h b;
#pragma unroll
  for (int i = 0; i < 16; ++i) b[i] = p[i];
  return b;
}

__device__ __forceinline__ float grp_max16(float x) {
#pragma unroll
  for (int m = 1; m < 16; m <<= 1) x = fmaxf(x, __shfl_xor(x, m, 32));
  return x;
}
__device__ __forceinline__ float grp_sum16(float x) {
#pragma unroll
  for (int m = 1; m < 16; m <<= 1) x += __shfl_xor(x, m, 32);
  return x;
}

// ---------------------------------------------------------------------------
// WMMA GEMM: out = act((A[M,K] @ Wt[N,K]^T + bias) * scale) + resid
// Wt is the TRANSPOSED weight (row n holds K contiguous halves).
// Block = 128 threads (4 waves) -> 16(M) x 64(N) tile; A tile + Wt tile are
// staged through LDS (async global->LDS when available).
// ---------------------------------------------------------------------------
template <bool HAS_BIAS, bool HAS_RES, bool RELU, bool WF32, bool WF16>
__global__ void gemm_wmma_kernel(const half_t* __restrict__ A, int lda,
                                 const half_t* __restrict__ Wt,
                                 const float* __restrict__ bias,
                                 const float* __restrict__ resid,
                                 float* __restrict__ C,
                                 half_t* __restrict__ Ch, int ldc,
                                 int K, float scale) {
  __shared__ __attribute__((aligned(16))) half_t Wsm[64 * 40];  // 64 x 32 (+8)
  __shared__ __attribute__((aligned(16))) half_t Asm[16 * 40];  // 16 x 32 (+8)

  int tid  = threadIdx.x;
  int wave = tid >> 5;
  int lane = tid & 31;
  int row0 = blockIdx.y * 16;
  int n0   = blockIdx.x * 64;

  v8f acc = {};
  for (int k0 = 0; k0 < K; k0 += 32) {
    // cooperative stage: 128 W chunks + 32 A chunks of 32B each
#pragma unroll
    for (int c = tid; c < 160; c += 128) {
      if (c < 128) {
        int r = c >> 1, off = (c & 1) * 16;
        copy32B_to_lds(Wt + (size_t)(n0 + r) * K + k0 + off, Wsm + r * 40 + off);
      } else {
        int c2 = c - 128;
        int r = c2 >> 1, off = (c2 & 1) * 16;
        copy32B_to_lds(A + (size_t)(row0 + r) * lda + k0 + off,
                       Asm + r * 40 + off);
      }
    }
    if (k0 + 32 < K) {
      __builtin_prefetch(&A[(size_t)row0 * lda + k0 + 32], 0, 1);
      __builtin_prefetch(&Wt[(size_t)n0 * K + k0 + 32], 0, 1);
    }
    lds_copy_drain();
    __syncthreads();

    v16h a = load_a_frag(Asm, 40, 0, 0, 15);
    v16h b = load_b_frag_kcontig(Wsm, 40, wave * 16, 0);
    acc = wmma_f16(a, b, acc);
    __syncthreads();
  }

  int hgrp = lane >> 4;
  int n = n0 + wave * 16 + (lane & 15);
#pragma unroll
  for (int v = 0; v < 8; ++v) {
    int m = row0 + v + 8 * hgrp;
    float val = acc[v];
    if (HAS_BIAS) val += bias[n];
    val *= scale;
    if (RELU) val = fmaxf(val, 0.f);
    if (HAS_RES) val += resid[(size_t)m * ldc + n];
    if (WF32) C[(size_t)m * ldc + n] = val;
    if (WF16) Ch[(size_t)m * ldc + n] = (half_t)val;
  }
}

// ---------------------------------------------------------------------------
// Flash-style attention, one wave per (batch, head, 16-query tile).
// ---------------------------------------------------------------------------
__global__ void attention_kernel(const half_t* __restrict__ Qh,
                                 const half_t* __restrict__ Kh,
                                 const half_t* __restrict__ vT,
                                 const float* __restrict__ qx,
                                 const float* __restrict__ qy,
                                 const float* __restrict__ qp0,
                                 const float* __restrict__ qp1,
                                 half_t* __restrict__ Oh) {
  __shared__ __attribute__((aligned(16))) half_t pt[16 * 40];

  int it = blockIdx.x;  // query tile 0..64
  int hh = blockIdx.y;  // head
  int bb = blockIdx.z;  // batch
  int lane = threadIdx.x & 31;
  int hgrp = lane >> 4;
  int row0 = it * 16;
  size_t base   = ((size_t)bb * SEQ) * DIMD + hh * HDD;
  size_t basevt = ((size_t)bb * NHD + hh) * HDD * (size_t)SEQP;

  v16h qa0 = load_a_frag(Qh + base, DIMD, row0, 0, SEQ - 1);
  v16h qa1 = load_a_frag(Qh + base, DIMD, row0, 32, SEQ - 1);

  float mrow[8], lrow[8];
  v8f accO[4];
  v8f zero = {};
#pragma unroll
  for (int v = 0; v < 8; ++v) { mrow[v] = -1e30f; lrow[v] = 0.f; }
#pragma unroll
  for (int nt = 0; nt < 4; ++nt) accO[nt] = zero;

  for (int jb = 0; jb < NJB; ++jb) {
    int j0 = jb * 32;
    float stile[2][8];

#pragma unroll
    for (int st = 0; st < 2; ++st) {
      int jc0 = j0 + st * 16;
      // K^T fragment: per lane 16 contiguous halves along d at row j
      int jr = jc0 + (lane & 15);
      if (jr >= SEQ) jr = SEQ - 1;
      const half_t* kp = Kh + base + (size_t)jr * DIMD + ((lane >> 4) ? 16 : 0);
      v16h kb0, kb1;
#pragma unroll
      for (int i = 0; i < 16; ++i) { kb0[i] = kp[i]; kb1[i] = kp[32 + i]; }
      v8f s = zero;
      s = wmma_f16(qa0, kb0, s);
      s = wmma_f16(qa1, kb1, s);
#pragma unroll
      for (int v = 0; v < 8; ++v) {
        int j = jc0 + (lane & 15);
        int i = row0 + v + 8 * hgrp;
        float sv;
        if (j >= SEQ) {
          sv = -1e30f;
        } else {
          float bia = 0.f;
          if (i < SEQ) {
            size_t ri = (size_t)bb * SEQ + i;
            if (i < NED && j < NED)       bia = qp0[ri * NHD + hh];
            else if (i < NED || j < NED)  bia = qp1[ri * NHD + hh];
            else {
              int ip = i - NED, jp = j - NED;
              int xd = (jp & 31) - (ip & 31) + DSZ;   // [1,63]
              int yd = (jp >> 5) - (ip >> 5) + DSZ;
              size_t qb = ri * DIMD + hh * HDD;
              bia = qx[qb + xd] + qy[qb + yd];
            }
          }
          sv = s[v] + bia;
        }
        stile[st][v] = sv;
      }
    }

    // online softmax (row = v + 8*hgrp)
#pragma unroll
    for (int v = 0; v < 8; ++v) {
      float rmax = grp_max16(fmaxf(stile[0][v], stile[1][v]));
      float mnew = fmaxf(mrow[v], rmax);
      float alpha = __expf(mrow[v] - mnew);
      float p0 = __expf(stile[0][v] - mnew);
      float p1 = __expf(stile[1][v] - mnew);
      lrow[v] = lrow[v] * alpha + grp_sum16(p0 + p1);
      mrow[v] = mnew;
#pragma unroll
      for (int nt = 0; nt < 4; ++nt) accO[nt][v] *= alpha;
      stile[0][v] = p0;
      stile[1][v] = p1;
    }

    // transpose P from C-layout to A-layout through LDS
    __syncthreads();
#pragma unroll
    for (int st = 0; st < 2; ++st)
#pragma unroll
      for (int v = 0; v < 8; ++v)
        pt[(v + 8 * hgrp) * 40 + st * 16 + (lane & 15)] = (half_t)stile[st][v];
    __syncthreads();
    v16h pa = load_a_frag(pt, 40, 0, 0, 15);

    // P @ V via V^T (K-contiguous along j)
#pragma unroll
    for (int nt = 0; nt < 4; ++nt) {
      int d = nt * 16 + (lane & 15);
      const half_t* vp = vT + basevt + (size_t)d * SEQP + j0 +
                         ((lane >> 4) ? 16 : 0);
      v16h vb;
#pragma unroll
      for (int i = 0; i < 16; ++i) vb[i] = vp[i];
      accO[nt] = wmma_f16(pa, vb, accO[nt]);
    }
  }

#pragma unroll
  for (int nt = 0; nt < 4; ++nt)
#pragma unroll
    for (int v = 0; v < 8; ++v) {
      int i = row0 + v + 8 * hgrp;
      if (i < SEQ) {
        int d = nt * 16 + (lane & 15);
        Oh[base + (size_t)i * DIMD + d] = (half_t)(accO[nt][v] / lrow[v]);
      }
    }
}

// ---------------------------------------------------------------------------
// Elementwise / reduction kernels
// ---------------------------------------------------------------------------
__global__ void embed_kernel(const float* __restrict__ x,
                             const float* __restrict__ patches,
                             const float* __restrict__ We,
                             const float* __restrict__ be,
                             float* __restrict__ h) {
  int row = blockIdx.x;
  int b = row / SEQ, s = row % SEQ;
  if (s < NED) {
    for (int d = threadIdx.x; d < DIMD; d += blockDim.x)
      h[(size_t)row * DIMD + d] = x[((size_t)b * NED + s) * DIMD + d];
  } else {
    int pix = s - NED, r = pix >> 5, c = pix & 31;
    float p0 = patches[(((size_t)b * 3 + 0) * DSZ + r) * DSZ + c];
    float p1 = patches[(((size_t)b * 3 + 1) * DSZ + r) * DSZ + c];
    float p2 = patches[(((size_t)b * 3 + 2) * DSZ + r) * DSZ + c];
    for (int d = threadIdx.x; d < DIMD; d += blockDim.x)
      h[(size_t)row * DIMD + d] =
          p0 * We[d] + p1 * We[DIMD + d] + p2 * We[2 * DIMD + d] + be[d];
  }
}

__global__ void ln_kernel(const float* __restrict__ X, const float* __restrict__ g,
                          const float* __restrict__ bt, half_t* __restrict__ Y) {
  int row = blockIdx.x;
  const float* xr = X + (size_t)row * DIMD;
  float s = 0.f, s2 = 0.f;
  for (int d = threadIdx.x; d < DIMD; d += blockDim.x) {
    float v = xr[d];
    s += v;
    s2 += v * v;
  }
#pragma unroll
  for (int m = 1; m < 32; m <<= 1) {
    s += __shfl_xor(s, m, 32);
    s2 += __shfl_xor(s2, m, 32);
  }
  __shared__ float sh1[8], sh2[8];
  int wave = threadIdx.x >> 5, lane = threadIdx.x & 31;
  if (lane == 0) { sh1[wave] = s; sh2[wave] = s2; }
  __syncthreads();
  int nw = blockDim.x >> 5;
  float ts = 0.f, ts2 = 0.f;
  for (int w = 0; w < nw; ++w) { ts += sh1[w]; ts2 += sh2[w]; }
  float mean = ts / DIMD;
  float var = ts2 / DIMD - mean * mean;
  float rinv = rsqrtf(var + 1e-5f);
  for (int d = threadIdx.x; d < DIMD; d += blockDim.x)
    Y[(size_t)row * DIMD + d] = (half_t)((xr[d] - mean) * rinv * g[d] + bt[d]);
}

__global__ void qpad_kernel(const half_t* __restrict__ Qh,
                            const float* __restrict__ pad0,
                            const float* __restrict__ pad1,
                            float* __restrict__ qp0, float* __restrict__ qp1) {
  int tid = blockIdx.x * blockDim.x + threadIdx.x;
  if (tid >= MROWS * NHD) return;
  int hh = tid % NHD;
  int row = tid / NHD;
  size_t base = (size_t)row * DIMD + hh * HDD;
  float a0 = 0.f, a1 = 0.f;
  for (int d = 0; d < HDD; ++d) {
    float q = (float)Qh[base + d];
    a0 += q * pad0[d];
    a1 += q * pad1[d];
  }
  qp0[tid] = a0;
  qp1[tid] = a1;
}

__global__ void classifier_kernel(const float* __restrict__ h,
                                  const float* __restrict__ Wc,
                                  const float* __restrict__ bc,
                                  float* __restrict__ out) {
  int tid = blockIdx.x * blockDim.x + threadIdx.x;
  if (tid >= BATCH * DSZ * DSZ * NCD) return;
  int nc = tid & 1;
  int pix = (tid >> 1) & (DSZ * DSZ - 1);
  int b = tid >> 11;
  const float* hr = h + ((size_t)b * SEQ + NED + pix) * DIMD;
  float acc = bc[nc];
  for (int d = 0; d < DIMD; ++d) acc += hr[d] * Wc[d * NCD + nc];
  out[((size_t)b * NCD + nc) * (DSZ * DSZ) + pix] = acc;
}

__global__ void f32_to_f16_kernel(const float* __restrict__ src,
                                  half_t* __restrict__ dst, int n) {
  int i = blockIdx.x * blockDim.x + threadIdx.x;
  if (i < n) dst[i] = (half_t)src[i];
}

// [K,N] f32 -> [N,K] f16 (weight transpose for K-contiguous B fragments)
__global__ void transpose_cvt_kernel(const float* __restrict__ src,
                                     half_t* __restrict__ dst, int K, int N) {
  int i = blockIdx.x * blockDim.x + threadIdx.x;
  if (i < K * N) {
    int k = i / N, n = i % N;
    dst[(size_t)n * K + k] = (half_t)src[i];
  }
}

// vH [B,S,DIM] f16 -> vT [B,NH,HD,SEQP] f16 (zero-padded tail)
__global__ void transpose_v_kernel(const half_t* __restrict__ vH,
                                   half_t* __restrict__ vT) {
  int tid = blockIdx.x * blockDim.x + threadIdx.x;
  const int total = BATCH * NHD * HDD * SEQP;
  if (tid >= total) return;
  int j = tid % SEQP;
  int r = tid / SEQP;
  int d = r % HDD;
  int r2 = r / HDD;
  int hh = r2 % NHD;
  int bb = r2 / NHD;
  half_t val = (half_t)0.f;
  if (j < SEQ) val = vH[((size_t)bb * SEQ + j) * DIMD + hh * HDD + d];
  vT[tid] = val;
}

// ---------------------------------------------------------------------------
// Host orchestration
// ---------------------------------------------------------------------------
extern "C" void kernel_launch(void* const* d_in, const int* in_sizes, int n_in,
                              void* d_out, int out_size, void* d_ws,
                              size_t ws_size, hipStream_t stream) {
  const float* x     = (const float*)d_in[0];
  const float* patch = (const float*)d_in[1];
  const float* We    = (const float*)d_in[2];
  const float* be    = (const float*)d_in[3];
  const float* x_emb = (const float*)d_in[4];
  const float* y_emb = (const float*)d_in[5];
  const float* pad0  = (const float*)d_in[6];
  const float* pad1  = (const float*)d_in[7];
  const float* Wq    = (const float*)d_in[8];
  const float* bq    = (const float*)d_in[9];
  const float* Wk    = (const float*)d_in[10];
  const float* bk    = (const float*)d_in[11];
  const float* Wv    = (const float*)d_in[12];
  const float* bv    = (const float*)d_in[13];
  const float* Wo    = (const float*)d_in[14];
  const float* bo    = (const float*)d_in[15];
  const float* ln1g  = (const float*)d_in[16];
  const float* ln1b  = (const float*)d_in[17];
  const float* ln2g  = (const float*)d_in[18];
  const float* ln2b  = (const float*)d_in[19];
  const float* W1    = (const float*)d_in[20];
  const float* b1    = (const float*)d_in[21];
  const float* W2    = (const float*)d_in[22];
  const float* b2    = (const float*)d_in[23];
  const float* Wc    = (const float*)d_in[24];
  const float* bc    = (const float*)d_in[25];
  float* out = (float*)d_out;
  (void)in_sizes; (void)n_in; (void)out_size; (void)ws_size;

  char* ws = (char*)d_ws;
  size_t off = 0;
  auto alloc = [&](size_t bytes) -> void* {
    void* p = (void*)(ws + off);
    off += (bytes + 255) & ~(size_t)255;
    return p;
  };

  float*  h   = (float*)alloc(4ull * MROWS * DIMD);
  half_t* hnH = (half_t*)alloc(2ull * MROWS * DIMD);
  half_t* qH  = (half_t*)alloc(2ull * MROWS * DIMD);
  half_t* kH  = (half_t*)alloc(2ull * MROWS * DIMD);
  half_t* vH  = (half_t*)alloc(2ull * MROWS * DIMD);
  half_t* vT  = (half_t*)alloc(2ull * BATCH * NHD * HDD * SEQP);
  half_t* oH  = (half_t*)alloc(2ull * MROWS * DIMD);
  half_t* ffH = (half_t*)alloc(2ull * MROWS * FFD);
  float*  qxB = (float*)alloc(4ull * MROWS * DIMD);
  float*  qyB = (float*)alloc(4ull * MROWS * DIMD);
  float*  qp0 = (float*)alloc(4ull * MROWS * NHD);
  float*  qp1 = (float*)alloc(4ull * MROWS * NHD);
  half_t* WqT = (half_t*)alloc(2ull * LAYERS * DIMD * DIMD);
  half_t* WkT = (half_t*)alloc(2ull * LAYERS * DIMD * DIMD);
  half_t* WvT = (half_t*)alloc(2ull * LAYERS * DIMD * DIMD);
  half_t* WoT = (half_t*)alloc(2ull * LAYERS * DIMD * DIMD);
  half_t* W1T = (half_t*)alloc(2ull * LAYERS * DIMD * FFD);
  half_t* W2T = (half_t*)alloc(2ull * LAYERS * FFD * DIMD);
  half_t* xT  = (half_t*)alloc(2ull * 64 * HDD);
  half_t* yT  = (half_t*)alloc(2ull * 64 * HDD);

  // weight transposes ([K,N] -> [N,K] f16)
  for (int l = 0; l < LAYERS; ++l) {
    int nsq = DIMD * DIMD, gsq = (nsq + 255) / 256;
    transpose_cvt_kernel<<<gsq, 256, 0, stream>>>(Wq + (size_t)l * nsq,
                                                  WqT + (size_t)l * nsq, DIMD, DIMD);
    transpose_cvt_kernel<<<gsq, 256, 0, stream>>>(Wk + (size_t)l * nsq,
                                                  WkT + (size_t)l * nsq, DIMD, DIMD);
    transpose_cvt_kernel<<<gsq, 256, 0, stream>>>(Wv + (size_t)l * nsq,
                                                  WvT + (size_t)l * nsq, DIMD, DIMD);
    transpose_cvt_kernel<<<gsq, 256, 0, stream>>>(Wo + (size_t)l * nsq,
                                                  WoT + (size_t)l * nsq, DIMD, DIMD);
    int nff = DIMD * FFD, gff = (nff + 255) / 256;
    transpose_cvt_kernel<<<gff, 256, 0, stream>>>(W1 + (size_t)l * nff,
                                                  W1T + (size_t)l * nff, DIMD, FFD);
    transpose_cvt_kernel<<<gff, 256, 0, stream>>>(W2 + (size_t)l * nff,
                                                  W2T + (size_t)l * nff, FFD, DIMD);
  }
  // x_emb/y_emb: logical W = emb^T [HD,64]; transposed layout == emb itself
  f32_to_f16_kernel<<<(64 * HDD + 255) / 256, 256, 0, stream>>>(x_emb, xT, 64 * HDD);
  f32_to_f16_kernel<<<(64 * HDD + 255) / 256, 256, 0, stream>>>(y_emb, yT, 64 * HDD);

  embed_kernel<<<MROWS, 128, 0, stream>>>(x, patch, We, be, h);

  const dim3 gQKV(DIMD / 64, MROWS / 16);
  const dim3 gEMB(64 / 64, MROWS / 16);
  const dim3 gFF1(FFD / 64, MROWS / 16);
  const float scl = 0.125f;  // 1/sqrt(HD)

  for (int l = 0; l < LAYERS; ++l) {
    ln_kernel<<<MROWS, 256, 0, stream>>>(h, ln1g + l * DIMD, ln1b + l * DIMD, hnH);
    // bias, ->f16
    gemm_wmma_kernel<true, false, false, false, true><<<gQKV, 128, 0, stream>>>(
        hnH, DIMD, WqT + (size_t)l * DIMD * DIMD, bq + l * DIMD, nullptr,
        nullptr, qH, DIMD, DIMD, scl);
    gemm_wmma_kernel<true, false, false, false, true><<<gQKV, 128, 0, stream>>>(
        hnH, DIMD, WkT + (size_t)l * DIMD * DIMD, bk + l * DIMD, nullptr,
        nullptr, kH, DIMD, DIMD, 1.f);
    gemm_wmma_kernel<true, false, false, false, true><<<gQKV, 128, 0, stream>>>(
        hnH, DIMD, WvT + (size_t)l * DIMD * DIMD, bv + l * DIMD, nullptr,
        nullptr, vH, DIMD, DIMD, 1.f);
    transpose_v_kernel<<<(BATCH * NHD * HDD * SEQP + 255) / 256, 256, 0,
                         stream>>>(vH, vT);
    // per-head q @ emb^T -> rel-pos dot tables (no bias, ->f32)
    for (int hh = 0; hh < NHD; ++hh) {
      gemm_wmma_kernel<false, false, false, true, false><<<gEMB, 128, 0, stream>>>(
          qH + hh * HDD, DIMD, xT, nullptr, nullptr, qxB + hh * HDD, nullptr,
          DIMD, HDD, 1.f);
      gemm_wmma_kernel<false, false, false, true, false><<<gEMB, 128, 0, stream>>>(
          qH + hh * HDD, DIMD, yT, nullptr, nullptr, qyB + hh * HDD, nullptr,
          DIMD, HDD, 1.f);
    }
    qpad_kernel<<<(MROWS * NHD + 255) / 256, 256, 0, stream>>>(qH, pad0, pad1,
                                                               qp0, qp1);
    dim3 ga((SEQ + 15) / 16, NHD, BATCH);
    attention_kernel<<<ga, 32, 0, stream>>>(qH, kH, vT, qxB, qyB, qp0, qp1, oH);
    // O proj: bias + residual, ->f32 (h)
    gemm_wmma_kernel<true, true, false, true, false><<<gQKV, 128, 0, stream>>>(
        oH, DIMD, WoT + (size_t)l * DIMD * DIMD, bo + l * DIMD, h, h, nullptr,
        DIMD, DIMD, 1.f);
    ln_kernel<<<MROWS, 256, 0, stream>>>(h, ln2g + l * DIMD, ln2b + l * DIMD, hnH);
    // FFN1: bias + relu, ->f16
    gemm_wmma_kernel<true, false, true, false, true><<<gFF1, 128, 0, stream>>>(
        hnH, DIMD, W1T + (size_t)l * DIMD * FFD, b1 + l * FFD, nullptr, nullptr,
        ffH, FFD, DIMD, 1.f);
    // FFN2: bias + residual, ->f32 (h)
    gemm_wmma_kernel<true, true, false, true, false><<<gQKV, 128, 0, stream>>>(
        ffH, FFD, W2T + (size_t)l * FFD * DIMD, b2 + l * DIMD, h, h, nullptr,
        DIMD, FFD, 1.f);
  }

  classifier_kernel<<<(BATCH * DSZ * DSZ * NCD + 255) / 256, 256, 0, stream>>>(
      h, Wc, bc, out);
}
